// Qwen3NextGatedDeltaNetPrefill_29497835389590
// MI455X (gfx1250) — matrix operand order repaired and checked by reading, not compiled, not run on hardware
//
#include <hip/hip_runtime.h>

// ---------------------------------------------------------------- constants
#define T_TOK 4096
#define HK 16
#define HV 32
#define DK 128
#define DV 128
#define KW 4
#define QKV (2 * HK * DK + HV * DV)   // 8192
#define CHUNK 64
#define NCHUNK (T_TOK / CHUNK)        // 64
#define DVS 32                        // dv slice per workgroup
#define NSLICE (DV / DVS)             // 4

typedef __bf16 bf16;
typedef bf16 v16bf __attribute__((ext_vector_type(16)));
typedef float v8f __attribute__((ext_vector_type(8)));
typedef int v4i __attribute__((ext_vector_type(4)));

typedef __attribute__((address_space(1))) v4i* gp_b128;  // global b128 pointer
typedef __attribute__((address_space(3))) v4i* lp_b128;  // LDS b128 pointer

#define WMMA_BF16(A, B, C) \
  __builtin_amdgcn_wmma_f32_16x16x32_bf16(false, (A), false, (B), (short)0, (C), false, false)

// ---------------------------------------------------------------- async copy helpers
__device__ __forceinline__ void async_b128_to_lds(const float* gsrc, void* ldst) {
#if __has_builtin(__builtin_amdgcn_global_load_async_to_lds_b128)
  __builtin_amdgcn_global_load_async_to_lds_b128((gp_b128)gsrc, (lp_b128)ldst, 0, 0);
#else
  unsigned loff = (unsigned)(size_t)(__attribute__((address_space(3))) char*)ldst;
  asm volatile("global_load_async_to_lds_b128 %0, %1, off" ::"v"(loff), "v"(gsrc) : "memory");
#endif
}

__device__ __forceinline__ void async_wait0() {
#if __has_builtin(__builtin_amdgcn_s_wait_asynccnt)
  __builtin_amdgcn_s_wait_asynccnt(0);
#else
  asm volatile("s_wait_asynccnt 0" ::: "memory");
#endif
}

// ---------------------------------------------------------------- operand gathers
// A operand (16x32 bf16, M=rows): lane holds row (lane&15); K runs [lhi*8,+8) and [16+lhi*8,+8)
__device__ __forceinline__ v16bf a_rows_bf(const bf16* __restrict__ p, int stride, int r,
                                           int k0, int lhi) {
  const bf16* row = p + (size_t)r * stride;
  int kb = k0 + lhi * 8;
  v16bf a;
#pragma unroll
  for (int i = 0; i < 8; ++i) a[i] = row[kb + i];
#pragma unroll
  for (int i = 0; i < 8; ++i) a[i + 8] = row[kb + 16 + i];
  return a;
}

__device__ __forceinline__ v16bf a_rows_bf_scaled(const bf16* __restrict__ p, int stride, int r,
                                                  int k0, int lhi, float sc) {
  const bf16* row = p + (size_t)r * stride;
  int kb = k0 + lhi * 8;
  v16bf a;
#pragma unroll
  for (int i = 0; i < 8; ++i) a[i] = (bf16)((float)row[kb + i] * sc);
#pragma unroll
  for (int i = 0; i < 8; ++i) a[i + 8] = (bf16)((float)row[kb + 16 + i] * sc);
  return a;
}

// A operand for K^T (rows = dk, K = token) with per-token decay scale exp(B[63]-B[j])
__device__ __forceinline__ v16bf a_kT_scaled(const bf16* __restrict__ p, int stride, int dk_row,
                                             int j0, int lhi, const float* Bc) {
  int jb = j0 + lhi * 8;
  v16bf a;
#pragma unroll
  for (int i = 0; i < 8; ++i) {
    int j = jb + i;
    a[i] = (bf16)((float)p[(size_t)j * stride + dk_row] * __expf(Bc[CHUNK - 1] - Bc[j]));
  }
#pragma unroll
  for (int i = 0; i < 8; ++i) {
    int j = jb + 16 + i;
    a[i + 8] = (bf16)((float)p[(size_t)j * stride + dk_row] * __expf(Bc[CHUNK - 1] - Bc[j]));
  }
  return a;
}

// B operand (32x16 bf16): lane holds col (lane&15); K runs [k0+lhi*8,+8) and [k0+16+lhi*8,+8).
// Source is row-major [N][K] (i.e. B = source^T), so contiguous along K.
__device__ __forceinline__ v16bf b_cols_bf(const bf16* __restrict__ p, int stride, int n,
                                           int k0, int lhi) {
  const bf16* row = p + (size_t)n * stride;
  int kb = k0 + lhi * 8;
  v16bf b;
#pragma unroll
  for (int i = 0; i < 16; ++i) b[i] = row[kb + (i & 7) + ((i >> 3) << 4)];
  return b;
}

// B operand from f32 [K][N] (row stride rstride), convert to bf16 on the fly
__device__ __forceinline__ v16bf b_f32_col(const float* __restrict__ p, int rstride, int n,
                                           int k0, int lhi) {
  const float* q = p + n;
  int kb = k0 + lhi * 8;
  v16bf b;
#pragma unroll
  for (int i = 0; i < 16; ++i) b[i] = (bf16)q[(size_t)(kb + (i & 7) + ((i >> 3) << 4)) * rstride];
  return b;
}

// ---------------------------------------------------------------- kernel 1a: conv+silu for V
__global__ void k_conv_v(const float* __restrict__ x, const float* __restrict__ w,
                         float* __restrict__ vbuf) {
  int t = blockIdx.x;
  for (int c = threadIdx.x; c < HV * DV; c += blockDim.x) {
    int ch = 2 * HK * DK + c;
    float acc = 0.f;
#pragma unroll
    for (int j = 0; j < KW; ++j) {
      int tt = t + j - (KW - 1);
      if (tt >= 0) acc += w[ch * KW + j] * x[(size_t)tt * QKV + ch];
    }
    vbuf[(size_t)t * (HV * DV) + c] = acc / (1.f + __expf(-acc));
  }
}

// ---------------------------------------------------------------- kernel 1b: conv+silu+l2norm for Q/K
__global__ void k_qk_norm(const float* __restrict__ x, const float* __restrict__ w,
                          bf16* __restrict__ qg, bf16* __restrict__ kg) {
  int t = blockIdx.x;
  int head = blockIdx.y;  // 0..15 -> q heads, 16..31 -> k heads
  int d = threadIdx.x;    // 0..127
  int ch = head * DK + d;
  float acc = 0.f;
#pragma unroll
  for (int j = 0; j < KW; ++j) {
    int tt = t + j - (KW - 1);
    if (tt >= 0) acc += w[ch * KW + j] * x[(size_t)tt * QKV + ch];
  }
  float sv = acc / (1.f + __expf(-acc));
  __shared__ float red[DK];
  red[d] = sv * sv;
  __syncthreads();
#pragma unroll
  for (int off = 64; off > 0; off >>= 1) {
    if (d < off) red[d] += red[d + off];
    __syncthreads();
  }
  float sc = rsqrtf(red[0] + 1e-6f);
  if (head < HK)
    qg[((size_t)t * HK + head) * DK + d] = (bf16)(sv * sc * 0.08838834764831845f);  // *DK^-0.5
  else
    kg[((size_t)t * HK + (head - HK)) * DK + d] = (bf16)(sv * sc);
}

// ---------------------------------------------------------------- kernel 1c: gating
__global__ void k_gating(const float* __restrict__ a, const float* __restrict__ b,
                         const float* __restrict__ dt_bias, const float* __restrict__ alog,
                         float* __restrict__ g, float* __restrict__ beta) {
  int i = blockIdx.x * blockDim.x + threadIdx.x;
  if (i >= T_TOK * HV) return;
  int h = i % HV;
  float xx = a[i] + dt_bias[h];
  float sp = (xx > 20.f) ? xx : log1pf(__expf(xx));
  g[i] = -__expf(alog[h]) * sp;
  beta[i] = 1.f / (1.f + __expf(-b[i]));
}

// ---------------------------------------------------------------- kernel 2: chunked gated delta rule
__launch_bounds__(256, 1)
__global__ void k_gdn(const bf16* __restrict__ qg, const bf16* __restrict__ kg,
                      const float* __restrict__ vbuf, const float* __restrict__ gG,
                      const float* __restrict__ betaG, float* __restrict__ out) {
  const int h = blockIdx.x;   // value head
  const int sl = blockIdx.y;  // dv slice
  const int kh = h >> 1;      // key head (GQA group = 2)
  const int c0 = sl * DVS;

  const int tid = threadIdx.x;
  const int lane = tid & 31;
  const int wave = tid >> 5;  // 8 waves
  const int lrow = lane & 15;
  const int lhi = lane >> 4;

  __shared__ float S[DK][DVS + 1];                      // fp32 state (persistent)
  __shared__ float Amat[CHUNK][CHUNK + 1];              // solve matrix
  __shared__ float Dm[CHUNK][DVS + 1];                  // RHS -> Delta (in place, f32)
  __shared__ __align__(16) bf16 QKb[CHUNK][CHUNK + 8];  // masked decayed QK^T
  __shared__ __align__(16) float Vst[CHUNK][DVS];       // async-staged V slice
  __shared__ __align__(16) bf16 Dbt[DVS][CHUNK + 8];    // Delta^T bf16 (B operand)
  __shared__ float Bc[CHUNK];                           // inclusive cumsum of g
  __shared__ float betac[CHUNK];

  for (int i = tid; i < DK * (DVS + 1); i += 256) (&S[0][0])[i] = 0.f;

  const int khstride = HK * DK;
  const float* vslice = vbuf + (size_t)h * DV + c0;  // + t*HV*DV + c

  // prologue: stage chunk 0's V slice into LDS (8 x 16-byte pieces per row)
  for (int f = tid; f < CHUNK * 8; f += 256) {
    int row = f >> 3, part = f & 7;
    async_b128_to_lds(vslice + (size_t)row * (HV * DV) + part * 4, &Vst[row][part * 4]);
  }
  __syncthreads();

  for (int ch = 0; ch < NCHUNK; ++ch) {
    const int t0 = ch * CHUNK;
    const bf16* kcp = kg + ((size_t)t0 * HK + kh) * DK;
    const bf16* qcp = qg + ((size_t)t0 * HK + kh) * DK;

    // prefetch next chunk's q/k (global_prefetch_b8)
    if (ch + 1 < NCHUNK && tid < CHUNK) {
      __builtin_prefetch(kcp + (size_t)(CHUNK + tid) * khstride, 0, 1);
      __builtin_prefetch(qcp + (size_t)(CHUNK + tid) * khstride, 0, 1);
    }

    // ---- phase 1: per-token scalars; wait for staged V; RHS = beta*v
    if (tid < CHUNK) betac[tid] = betaG[(size_t)(t0 + tid) * HV + h];
    if (tid == 0) {
      float acc = 0.f;
      for (int t = 0; t < CHUNK; ++t) {
        acc += gG[(size_t)(t0 + t) * HV + h];
        Bc[t] = acc;
      }
    }
    async_wait0();    // own wave's async loads complete
    __syncthreads();  // everyone's complete; scalars visible
    for (int i = tid; i < CHUNK * DVS; i += 256) {
      int t = i / DVS, c = i % DVS;
      Dm[t][c] = betac[t] * Vst[t][c];
    }
    __syncthreads();
    // V staging buffer consumed -> kick off async load of next chunk's V slice
    if (ch + 1 < NCHUNK) {
      for (int f = tid; f < CHUNK * 8; f += 256) {
        int row = f >> 3, part = f & 7;
        async_b128_to_lds(vslice + (size_t)(t0 + CHUNK + row) * (HV * DV) + part * 4,
                          &Vst[row][part * 4]);
      }
    }

    // ---- phase 2a: A = strict_lower( beta_t * exp(B_t - B_j) * (k_t . k_j) )
    for (int it = 0; it < 2; ++it) {
      int tile = wave * 2 + it;  // 16 tiles over 64x64
      int mt = tile >> 2, nt = tile & 3;
      v8f acc = {};
#pragma unroll
      for (int kk = 0; kk < DK; kk += 32) {
        v16bf A = a_rows_bf(kcp, khstride, mt * 16 + lrow, kk, lhi);
        v16bf B = b_cols_bf(kcp, khstride, nt * 16 + lrow, kk, lhi);
        acc = WMMA_BF16(A, B, acc);
      }
#pragma unroll
      for (int r = 0; r < 8; ++r) {
        int tr = mt * 16 + r + 8 * lhi, tc = nt * 16 + lrow;
        Amat[tr][tc] = (tc < tr) ? betac[tr] * __expf(Bc[tr] - Bc[tc]) * acc[r] : 0.f;
      }
    }
    // ---- phase 2b: RHS -= (beta_t * exp(B_t) * k_t) @ S0
    {
      int mt = wave & 3, nt = wave >> 2;  // 8 tiles over 64x32
      int trl = mt * 16 + lrow;
      float sc = betac[trl] * __expf(Bc[trl]);
      v8f acc = {};
#pragma unroll
      for (int kk = 0; kk < DK; kk += 32) {
        v16bf A = a_rows_bf_scaled(kcp, khstride, trl, kk, lhi, sc);
        v16bf B = b_f32_col(&S[0][0], DVS + 1, nt * 16 + lrow, kk, lhi);
        acc = WMMA_BF16(A, B, acc);
      }
#pragma unroll
      for (int r = 0; r < 8; ++r) Dm[mt * 16 + r + 8 * lhi][nt * 16 + lrow] -= acc[r];
    }
    __syncthreads();

    // ---- phase 3: forward substitution (I + A) Delta = RHS, in place; columns independent.
    //      Also emit Delta^T as bf16 (contiguous per-lane rows) for later B operands.
    if (wave == 0) {
      int c = lane;  // DVS == 32 columns, one per lane
      Dbt[c][0] = (bf16)Dm[0][c];
      for (int t = 1; t < CHUNK; ++t) {
        float s = Dm[t][c];
        for (int j = 0; j < t; ++j) s -= Amat[t][j] * Dm[j][c];
        Dm[t][c] = s;
        Dbt[c][t] = (bf16)s;
      }
    }
    __syncthreads();

    // ---- phase 4: masked decayed QK^T (incl. diagonal) -> bf16
    for (int it = 0; it < 2; ++it) {
      int tile = wave * 2 + it;
      int mt = tile >> 2, nt = tile & 3;
      v8f acc = {};
#pragma unroll
      for (int kk = 0; kk < DK; kk += 32) {
        v16bf A = a_rows_bf(qcp, khstride, mt * 16 + lrow, kk, lhi);
        v16bf B = b_cols_bf(kcp, khstride, nt * 16 + lrow, kk, lhi);
        acc = WMMA_BF16(A, B, acc);
      }
#pragma unroll
      for (int r = 0; r < 8; ++r) {
        int tr = mt * 16 + r + 8 * lhi, tc = nt * 16 + lrow;
        QKb[tr][tc] = (bf16)((tc <= tr) ? __expf(Bc[tr] - Bc[tc]) * acc[r] : 0.f);
      }
    }
    __syncthreads();

    // ---- phase 5: O = (exp(B_t) q_t) @ S0 + (QK masked) @ Delta
    {
      int mt = wave & 3, nt = wave >> 2;
      int trl = mt * 16 + lrow;
      float qs = __expf(Bc[trl]);
      v8f acc = {};
#pragma unroll
      for (int kk = 0; kk < DK; kk += 32) {
        v16bf A = a_rows_bf_scaled(qcp, khstride, trl, kk, lhi, qs);
        v16bf B = b_f32_col(&S[0][0], DVS + 1, nt * 16 + lrow, kk, lhi);
        acc = WMMA_BF16(A, B, acc);
      }
#pragma unroll
      for (int kk = 0; kk < CHUNK; kk += 32) {
        v16bf A = a_rows_bf(&QKb[0][0], CHUNK + 8, trl, kk, lhi);
        v16bf B = b_cols_bf(&Dbt[0][0], CHUNK + 8, nt * 16 + lrow, kk, lhi);
        acc = WMMA_BF16(A, B, acc);
      }
#pragma unroll
      for (int r = 0; r < 8; ++r) {
        int t = t0 + mt * 16 + r + 8 * lhi;
        out[(size_t)t * (HV * DV) + h * DV + c0 + nt * 16 + lrow] = acc[r];
      }
    }
    __syncthreads();

    // ---- phase 6: S = exp(B_last) * S + (exp(B_last - B_j) k_j)^T @ Delta
    {
      float gl = __expf(Bc[CHUNK - 1]);
      int mt = wave;  // 8 dk-tiles of 16
#pragma unroll
      for (int nt = 0; nt < 2; ++nt) {
        v8f acc;
#pragma unroll
        for (int r = 0; r < 8; ++r) acc[r] = S[mt * 16 + r + 8 * lhi][nt * 16 + lrow] * gl;
#pragma unroll
        for (int j0 = 0; j0 < CHUNK; j0 += 32) {
          v16bf A = a_kT_scaled(kcp, khstride, mt * 16 + lrow, j0, lhi, Bc);
          v16bf B = b_cols_bf(&Dbt[0][0], CHUNK + 8, nt * 16 + lrow, j0, lhi);
          acc = WMMA_BF16(A, B, acc);
        }
#pragma unroll
        for (int r = 0; r < 8; ++r) S[mt * 16 + r + 8 * lhi][nt * 16 + lrow] = acc[r];
      }
    }
    __syncthreads();
  }
}

// ---------------------------------------------------------------- host launch
extern "C" void kernel_launch(void* const* d_in, const int* in_sizes, int n_in,
                              void* d_out, int out_size, void* d_ws, size_t ws_size,
                              hipStream_t stream) {
  const float* mixed = (const float*)d_in[0];
  const float* a = (const float*)d_in[1];
  const float* b = (const float*)d_in[2];
  const float* conv_w = (const float*)d_in[3];
  const float* dt_bias = (const float*)d_in[4];
  const float* alog = (const float*)d_in[5];
  float* out = (float*)d_out;

  char* ws = (char*)d_ws;
  float* vbuf = (float*)ws;                                         // T*HV*DV f32 (64MB)
  bf16* qg = (bf16*)(ws + (size_t)T_TOK * HV * DV * 4);             // T*HK*DK bf16 (16MB)
  bf16* kg = (bf16*)((char*)qg + (size_t)T_TOK * HK * DK * 2);      // (16MB)
  float* gbuf = (float*)((char*)kg + (size_t)T_TOK * HK * DK * 2);  // T*HV f32
  float* betabuf = gbuf + (size_t)T_TOK * HV;

  k_conv_v<<<T_TOK, 256, 0, stream>>>(mixed, conv_w, vbuf);
  dim3 gqk(T_TOK, 2 * HK);
  k_qk_norm<<<gqk, DK, 0, stream>>>(mixed, conv_w, qg, kg);
  k_gating<<<(T_TOK * HV + 255) / 256, 256, 0, stream>>>(a, b, dt_bias, alog, gbuf, betabuf);
  dim3 gmain(HV, NSLICE);
  k_gdn<<<gmain, 256, 0, stream>>>(qg, kg, vbuf, gbuf, betabuf, out);
}
